// SpatialGraphDiffusion_17025250361515
// MI455X (gfx1250) — matrix operand order repaired and checked by reading, not compiled, not run on hardware
//
#include <hip/hip_runtime.h>
#include <math.h>

#define N_NODES 1024
#define HDIM 64
#define NLAYERS 3

typedef float    v2f  __attribute__((ext_vector_type(2)));
typedef float    v8f  __attribute__((ext_vector_type(8)));
typedef _Float16 v16h __attribute__((ext_vector_type(16)));

union V8F  { v8f v;  float    e[8];  };
union V16H { v16h v; _Float16 e[16]; };

// ---- fast math helpers (raw HW trans ops, no IEEE fixup sequences) --------
__device__ __forceinline__ float fast_rcp(float x) {
#if __has_builtin(__builtin_amdgcn_rcpf)
  return __builtin_amdgcn_rcpf(x);      // v_rcp_f32
#else
  return 1.0f / x;
#endif
}
__device__ __forceinline__ float fast_sqrt(float x) {
#if __has_builtin(__builtin_amdgcn_sqrtf)
  return __builtin_amdgcn_sqrtf(x);     // v_sqrt_f32
#else
  return sqrtf(x);
#endif
}
__device__ __forceinline__ float sigm_f(float x) {
#if __has_builtin(__builtin_amdgcn_tanhf)
  return __builtin_amdgcn_tanhf(0.5f * x) * 0.5f + 0.5f;  // single v_tanh_f32
#else
  return fast_rcp(1.0f + __expf(-x));                      // exp + rcp
#endif
}
__device__ __forceinline__ float silu_f(float x) { return x * sigm_f(x); }

#if __has_builtin(__builtin_amdgcn_wmma_f32_16x16x4_f32)
#define HAVE_WMMA_K4 1
#else
#define HAVE_WMMA_K4 0
#endif

// ---------------------------------------------------------------------------
// 1) timestep embedding + time MLP (single block)
// ---------------------------------------------------------------------------
__global__ __launch_bounds__(64) void k_time_embed(
    const int* __restrict__ t_ptr,
    const float* __restrict__ tw1, const float* __restrict__ tb1,
    const float* __restrict__ tw2, const float* __restrict__ tb2,
    float* __restrict__ te_out) {
  __shared__ float te[64], t1[64];
  int n = threadIdx.x;
  float tval = (float)(*t_ptr);
  int k = n & 31;
  // -ln(10000)/31
  float freq = __expf((float)k * -0.29710775548f);
  float tf = tval * freq;
  te[n] = (n < 32) ? __sinf(tf) : __cosf(tf);
  __syncthreads();
  float acc = tb1[n];
  for (int c = 0; c < 64; ++c) acc += te[c] * tw1[c * 64 + n];
  t1[n] = silu_f(acc);
  __syncthreads();
  float acc2 = tb2[n];
  for (int c = 0; c < 64; ++c) acc2 += t1[c] * tw2[c * 64 + n];
  te_out[n] = acc2;
}

// ---------------------------------------------------------------------------
// 2) h0 = pos @ pw + pb + te
// ---------------------------------------------------------------------------
__global__ __launch_bounds__(256) void k_h0(
    const float* __restrict__ pos, const float* __restrict__ pw,
    const float* __restrict__ pb, const float* __restrict__ te,
    float* __restrict__ h0) {
  int idx = blockIdx.x * 256 + threadIdx.x;  // N*64 total
  int i = idx >> 6, n = idx & 63;
  float acc = pb[n] + te[n];
  acc += pos[i * 3 + 0] * pw[0 * 64 + n];
  acc += pos[i * 3 + 1] * pw[1 * 64 + n];
  acc += pos[i * 3 + 2] * pw[2 * 64 + n];
  h0[idx] = acc;
}

// ---------------------------------------------------------------------------
// 3) Heavy pass: S[l][i][:] = (sum_j adj[i,j]*silu(ef_in@ew1+eb1)) @ ew2
//                            + eb2 * rowsum(adj[i])
//    One wave per (i, l). GEMM1 (K=4) via V_WMMA_F32_16X16X4_F32 with the
//    eb1 bias folded into the C operand. Adjacency weights arrive as two
//    b128 half-wave broadcasts. Accumulation is split into even/odd channels
//    so (d[2r],d[2r+1]) x (w[2r],w[2r+1]) fuse into v_pk_fma_f32 without
//    cross-tile register shuffling.
// ---------------------------------------------------------------------------
__global__ __launch_bounds__(32) void k_edge_msg(
    const float* __restrict__ pos, const float* __restrict__ adj,
    const float* __restrict__ ew1, const float* __restrict__ eb1,
    const float* __restrict__ ew2, const float* __restrict__ eb2,
    float* __restrict__ S) {
  const int i = blockIdx.x;
  const int l = blockIdx.y;
  const int lane = threadIdx.x;
  const int col = lane & 15;
  const int hf = lane >> 4;  // 0: low rows/K half, 1: high half

  const float* W1 = ew1 + l * 4 * HDIM;
  const float* B1 = eb1 + l * HDIM;
  const float* W2 = ew2 + l * HDIM * HDIM;
  const float* B2 = eb2 + l * HDIM;

#if HAVE_WMMA_K4
  // B operand (4x16 f32): lanes0-15 hold K0/K1, lanes16-31 hold K2/K3
  v2f Bop[4];
#pragma unroll
  for (int t = 0; t < 4; ++t) {
    int n = t * 16 + col;
    Bop[t].x = W1[(hf * 2 + 0) * HDIM + n];
    Bop[t].y = W1[(hf * 2 + 1) * HDIM + n];
  }
  // C operand: bias eb1[n] broadcast over all rows (constant per lane)
  V8F Cb[4];
#pragma unroll
  for (int t = 0; t < 4; ++t) {
    float bn = B1[t * 16 + col];
#pragma unroll
    for (int r = 0; r < 8; ++r) Cb[t].e[r] = bn;
  }
#else
  float Wf[4][4];
#pragma unroll
  for (int t = 0; t < 4; ++t)
#pragma unroll
    for (int c = 0; c < 4; ++c) Wf[t][c] = W1[c * HDIM + t * 16 + col];
  float bias1[4];
#pragma unroll
  for (int t = 0; t < 4; ++t) bias1[t] = B1[t * 16 + col];
#endif

  const float pix = pos[i * 3 + 0];
  const float piy = pos[i * 3 + 1];
  const float piz = pos[i * 3 + 2];

  // adjacency row, viewed as float4; each lane reads the 8 weights of its
  // half's rows (same address across the 16 lanes of a half -> broadcast)
  const float4* arow = (const float4*)(adj + (size_t)i * N_NODES);
  const float rsel = (col == 0) ? 1.0f : 0.0f;  // dedup rowsum accumulation

  // even/odd accumulation channels per tile -> natural v_pk_fma pairs
  float accx[4] = {0.f, 0.f, 0.f, 0.f};
  float accy[4] = {0.f, 0.f, 0.f, 0.f};
  float rs = 0.0f;

  for (int jt = 0; jt < 64; ++jt) {
    int j = jt * 16 + col;  // lanes L and L+16 handle the same pair row
    float ex = pix - pos[j * 3 + 0];
    float ey = piy - pos[j * 3 + 1];
    float ez = piz - pos[j * 3 + 2];
    float ed = fast_sqrt(ex * ex + ey * ey + ez * ez);

    // weights for D rows hf*8 + 0..7
    float4 w0 = arow[jt * 4 + hf * 2 + 0];
    float4 w1 = arow[jt * 4 + hf * 2 + 1];
    rs += rsel * (((w0.x + w0.y) + (w0.z + w0.w)) +
                  ((w1.x + w1.y) + (w1.z + w1.w)));
    float w[8] = {w0.x, w0.y, w0.z, w0.w, w1.x, w1.y, w1.z, w1.w};

#if HAVE_WMMA_K4
    // A operand (16x4 f32): lanes0-15: K0=rel.x, K1=rel.y; lanes16-31: K2=rel.z, K3=dist
    v2f Aop;
    Aop.x = hf ? ez : ex;
    Aop.y = hf ? ed : ey;
#pragma unroll
    for (int t = 0; t < 4; ++t) {
      V8F d;
      d.v = __builtin_amdgcn_wmma_f32_16x16x4_f32(
          false, Aop, false, Bop[t], (short)0, Cb[t].v, false, false);
      // D layout: vgpr r -> row (r + 8*hf), lane%16 -> col; silu + adj-weight
#pragma unroll
      for (int r2 = 0; r2 < 4; ++r2) {
        accx[t] += w[2 * r2 + 0] * silu_f(d.e[2 * r2 + 0]);
        accy[t] += w[2 * r2 + 1] * silu_f(d.e[2 * r2 + 1]);
      }
    }
#else
    float exr[8], eyr[8], ezr[8], edr[8];
#pragma unroll
    for (int r = 0; r < 8; ++r) {
      int src = (hf << 3) + r;
      exr[r] = __shfl(ex, src, 32);
      eyr[r] = __shfl(ey, src, 32);
      ezr[r] = __shfl(ez, src, 32);
      edr[r] = __shfl(ed, src, 32);
    }
#pragma unroll
    for (int t = 0; t < 4; ++t)
#pragma unroll
      for (int r = 0; r < 8; ++r) {
        float y = exr[r] * Wf[t][0] + eyr[r] * Wf[t][1] +
                  ezr[r] * Wf[t][2] + edr[r] * Wf[t][3] + bias1[t];
        if (r & 1) accy[t] += w[r] * silu_f(y);
        else       accx[t] += w[r] * silu_f(y);
      }
#endif
  }

  // merge channels, then combine row halves: u[k] for k = t*16+col
  float acc[4];
#pragma unroll
  for (int t = 0; t < 4; ++t) acc[t] = accx[t] + accy[t];
#pragma unroll
  for (int t = 0; t < 4; ++t) acc[t] += __shfl_xor(acc[t], 16, 32);
  // rowsum across the wave (only col==0 lanes contributed)
#pragma unroll
  for (int m = 16; m >= 1; m >>= 1) rs += __shfl_xor(rs, m, 32);

  __shared__ float u[64];
  if (hf == 0) {
#pragma unroll
    for (int t = 0; t < 4; ++t) u[t * 16 + col] = acc[t];
  }
  __syncthreads();

  // GEMV: S[i] = u @ ew2 + eb2 * rowsum   (2 outputs per lane)
  float s0 = 0.f, s1 = 0.f;
  for (int k = 0; k < 64; ++k) {
    float uk = u[k];
    s0 += uk * W2[k * HDIM + lane];
    s1 += uk * W2[k * HDIM + lane + 32];
  }
  float* So = S + ((size_t)l * N_NODES + i) * HDIM;
  So[lane]      = s0 + B2[lane] * rs;
  So[lane + 32] = s1 + B2[lane + 32] * rs;
}

// ---------------------------------------------------------------------------
// 4) node update: h_out = silu(cat(h, h+S[l], te) @ mw[l] + mb[l])
//    1024x192x64 GEMM in f16 WMMA; mb folded into the accumulator init.
// ---------------------------------------------------------------------------
__global__ __launch_bounds__(32) void k_node_update(
    const float* __restrict__ h_in, const float* __restrict__ S,
    const float* __restrict__ te, const float* __restrict__ mw,
    const float* __restrict__ mb, float* __restrict__ h_out, int l) {
  const int blk = blockIdx.x;
  const int lane = threadIdx.x;
  const int col = lane & 15;
  const int hf = lane >> 4;
  const int row = blk * 16 + col;  // A: lanes 0-15 and 16-31 both map M=col
  const float* hrow = h_in + (size_t)row * HDIM;
  const float* srow = S + ((size_t)l * N_NODES + row) * HDIM;
  const float* W = mw + (size_t)l * 3 * HDIM * HDIM;
  const float* bb = mb + l * HDIM;

  V8F cacc[4];
#pragma unroll
  for (int t = 0; t < 4; ++t) {
    float bn = bb[t * 16 + col];
#pragma unroll
    for (int r = 0; r < 8; ++r) cacc[t].e[r] = bn;  // bias in accumulator
  }

  for (int kk = 0; kk < 6; ++kk) {
    // A operand: low 8 halves = K kk*32+hf*8+{0..7}, high 8 = +16
    V16H a;
#pragma unroll
    for (int q = 0; q < 16; ++q) {
      int c = kk * 32 + hf * 8 + (q & 7) + ((q >> 3) << 4);
      float xv;
      if (c < 64)        xv = hrow[c];
      else if (c < 128)  xv = hrow[c - 64] + srow[c - 64];   // msg = S + h
      else               xv = te[c - 128];
      a.e[q] = (_Float16)xv;
    }
#pragma unroll
    for (int t = 0; t < 4; ++t) {
      // B operand: low 8 halves = K kk*32+hf*8+{0..7}, high 8 = +16
      V16H b;
      int n = t * 16 + col;
      int kbase = kk * 32 + hf * 16;
#pragma unroll
      for (int q = 0; q < 16; ++q)
        b.e[q] = (_Float16)W[(size_t)(kk * 32 + hf * 8 + (q & 7) + ((q >> 3) << 4)) * HDIM + n];
      cacc[t].v = __builtin_amdgcn_wmma_f32_16x16x32_f16(
          false, a.v, false, b.v, (short)0, cacc[t].v, false, false);
    }
  }
#pragma unroll
  for (int t = 0; t < 4; ++t) {
    int n = t * 16 + col;
#pragma unroll
    for (int r = 0; r < 8; ++r) {
      int m = blk * 16 + hf * 8 + r;
      h_out[(size_t)m * HDIM + n] = silu_f(cacc[t].e[r]);
    }
  }
}

// ---------------------------------------------------------------------------
// 5) hi = h @ aw1[:H], hj = h @ aw1[H:]
// ---------------------------------------------------------------------------
__global__ __launch_bounds__(256) void k_hihj(
    const float* __restrict__ h, const float* __restrict__ aw1,
    float* __restrict__ hi_o, float* __restrict__ hj_o) {
  int idx = blockIdx.x * 256 + threadIdx.x;
  int i = idx >> 6, n = idx & 63;
  const float* hr = h + (size_t)i * 64;
  float a = 0.f, b = 0.f;
  for (int k = 0; k < 64; ++k) {
    float hv = hr[k];
    a += hv * aw1[k * 64 + n];
    b += hv * aw1[(64 + k) * 64 + n];
  }
  hi_o[idx] = a;
  hj_o[idx] = b;
}

// ---------------------------------------------------------------------------
// 6) adj head: symmetric sigmoid logits over upper-triangular 16x16 blocks
// ---------------------------------------------------------------------------
__global__ __launch_bounds__(256) void k_adj_head(
    const float* __restrict__ hi, const float* __restrict__ hj,
    const float* __restrict__ ab1, const float* __restrict__ aw2,
    const float* __restrict__ ab2, const float* __restrict__ mask,
    float* __restrict__ out) {
  int bi = blockIdx.x, bj = blockIdx.y;
  if (bj < bi) return;
  __shared__ float sHiI[16 * 64], sHjJ[16 * 64], sHiJ[16 * 64], sHjI[16 * 64];
  int t = threadIdx.x;
  for (int idx = t; idx < 1024; idx += 256) {
    int r = idx >> 6, c = idx & 63;
    sHiI[idx] = hi[(size_t)(bi * 16 + r) * 64 + c];
    sHjI[idx] = hj[(size_t)(bi * 16 + r) * 64 + c];
    sHiJ[idx] = hi[(size_t)(bj * 16 + r) * 64 + c];
    sHjJ[idx] = hj[(size_t)(bj * 16 + r) * 64 + c];
  }
  __syncthreads();
  int ti = t >> 4, tj = t & 15;
  int i = bi * 16 + ti, j = bj * 16 + tj;
  float a1 = 0.f, a2 = 0.f;
  for (int k = 0; k < 64; ++k) {
    float b = ab1[k], w = aw2[k];
    a1 += silu_f(sHiI[ti * 64 + k] + sHjJ[tj * 64 + k] + b) * w;  // logits[i,j]
    a2 += silu_f(sHiJ[tj * 64 + k] + sHjI[ti * 64 + k] + b) * w;  // logits[j,i]
  }
  float lg = 0.5f * (a1 + a2) + ab2[0];
  float p = sigm_f(lg) * mask[i] * mask[j];
  out[(size_t)i * N_NODES + j] = p;
  out[(size_t)j * N_NODES + i] = p;
}

// ---------------------------------------------------------------------------
// 7) eps head: eps = (silu(cat(h,te)@hw1+hb1) @ hw2 + hb2) * mask
// ---------------------------------------------------------------------------
__global__ __launch_bounds__(64) void k_eps_head(
    const float* __restrict__ h, const float* __restrict__ te,
    const float* __restrict__ hw1, const float* __restrict__ hb1,
    const float* __restrict__ hw2, const float* __restrict__ hb2,
    const float* __restrict__ mask, float* __restrict__ out) {
  int i = blockIdx.x, t = threadIdx.x;
  __shared__ float mid[64];
  const float* hr = h + (size_t)i * 64;
  float acc = hb1[t];
  for (int c = 0; c < 64; ++c) acc += hr[c] * hw1[c * 64 + t];
  for (int c = 0; c < 64; ++c) acc += te[c] * hw1[(64 + c) * 64 + t];
  mid[t] = silu_f(acc);
  __syncthreads();
  if (t < 3) {
    float e = hb2[t];
    for (int k = 0; k < 64; ++k) e += mid[k] * hw2[k * 3 + t];
    out[(size_t)N_NODES * N_NODES + (size_t)i * 3 + t] = e * mask[i];
  }
}

// ---------------------------------------------------------------------------
extern "C" void kernel_launch(void* const* d_in, const int* in_sizes, int n_in,
                              void* d_out, int out_size, void* d_ws,
                              size_t ws_size, hipStream_t stream) {
  (void)in_sizes; (void)n_in; (void)out_size; (void)ws_size;
  const float* pos  = (const float*)d_in[0];
  const float* adj  = (const float*)d_in[1];
  const float* mask = (const float*)d_in[2];
  const int*   tptr = (const int*)d_in[3];
  const float* tw1  = (const float*)d_in[4];
  const float* tb1  = (const float*)d_in[5];
  const float* tw2  = (const float*)d_in[6];
  const float* tb2  = (const float*)d_in[7];
  const float* pw   = (const float*)d_in[8];
  const float* pb   = (const float*)d_in[9];
  const float* ew1  = (const float*)d_in[10];
  const float* eb1  = (const float*)d_in[11];
  const float* ew2  = (const float*)d_in[12];
  const float* eb2  = (const float*)d_in[13];
  const float* mw   = (const float*)d_in[14];
  const float* mb   = (const float*)d_in[15];
  const float* aw1  = (const float*)d_in[16];
  const float* ab1  = (const float*)d_in[17];
  const float* aw2  = (const float*)d_in[18];
  const float* ab2  = (const float*)d_in[19];
  const float* hw1  = (const float*)d_in[20];
  const float* hb1  = (const float*)d_in[21];
  const float* hw2  = (const float*)d_in[22];
  const float* hb2  = (const float*)d_in[23];

  float* ws = (float*)d_ws;
  float* te = ws;                                  // 64
  float* hA = te + 64;                             // N*H
  float* hB = hA + N_NODES * HDIM;                 // N*H
  float* S  = hB + N_NODES * HDIM;                 // L*N*H
  float* hi = S + NLAYERS * N_NODES * HDIM;        // N*H
  float* hj = hi + N_NODES * HDIM;                 // N*H
  float* out = (float*)d_out;

  hipLaunchKernelGGL(k_time_embed, dim3(1), dim3(64), 0, stream,
                     tptr, tw1, tb1, tw2, tb2, te);
  hipLaunchKernelGGL(k_h0, dim3(N_NODES * HDIM / 256), dim3(256), 0, stream,
                     pos, pw, pb, te, hA);
  hipLaunchKernelGGL(k_edge_msg, dim3(N_NODES, NLAYERS), dim3(32), 0, stream,
                     pos, adj, ew1, eb1, ew2, eb2, S);
  float* hc = hA;
  float* hn = hB;
  for (int l = 0; l < NLAYERS; ++l) {
    hipLaunchKernelGGL(k_node_update, dim3(N_NODES / 16), dim3(32), 0, stream,
                       hc, S, te, mw, mb, hn, l);
    float* tmp = hc; hc = hn; hn = tmp;
  }
  hipLaunchKernelGGL(k_hihj, dim3(N_NODES * HDIM / 256), dim3(256), 0, stream,
                     hc, aw1, hi, hj);
  hipLaunchKernelGGL(k_adj_head, dim3(64, 64), dim3(256), 0, stream,
                     hi, hj, ab1, aw2, ab2, mask, out);
  hipLaunchKernelGGL(k_eps_head, dim3(N_NODES), dim3(64), 0, stream,
                     hc, te, hw1, hb1, hw2, hb2, mask, out);
}
